// NeighConsensus_29274497089741
// MI455X (gfx1250) — matrix operand order repaired and checked by reading, not compiled, not run on hardware
//
#include <hip/hip_runtime.h>

// CDNA5 / gfx1250, wave32. bf16 WMMA 16x16x32 with f32 accumulation.
typedef __attribute__((ext_vector_type(16))) __bf16 v16bf;
typedef __attribute__((ext_vector_type(8)))  float  v8f;

#define SDIM   32
#define CPAD   16
#define NTAP27 27
#define FRAGS  54          // 27 (k1,k2,k3) taps * 2 K-pairs (k4 in {0,1} / {2,pad})
#define FRAG_ELS 512       // 32 lanes * 16 bf16 per A fragment
#define NPOS   (4L*32*32*32*32)   // B*D1*D2*D3*D4 output positions
#define PROWS  36          // patch rows: 3(k1) * 3(k2) * 4(s3 halo for 2 d3)
#define PCHUNK (PROWS*34*2) // 16-byte fill chunks

__device__ __forceinline__ unsigned short f32_to_bf16(float f) {
  unsigned u = __float_as_uint(f);
  u += 0x7fffu + ((u >> 16) & 1u);           // round-to-nearest-even
  return (unsigned short)(u >> 16);
}

// ---------------------------------------------------------------------------
// Pre-pack conv weights [Cout,Cin,3,3,3,3] f32 into per-fragment bf16 layout
// with contiguous K per lane half (K = 16*(lane/16) + slot, as B uses), so the compute
// kernel loads each lane's A fragment as 32 contiguous bytes.
// ---------------------------------------------------------------------------
__global__ void nc_pack_w(const float* __restrict__ w,
                          unsigned short* __restrict__ wp,
                          int coutR, int cinR) {
  int idx = blockIdx.x * blockDim.x + threadIdx.x;
  if (idx >= FRAGS * FRAG_ELS) return;
  int frag = idx >> 9;           // 0..53
  int el   = idx & 511;
  int lane = el >> 4;            // 0..31
  int slot = el & 15;            // VGPR*2 + half
  int v  = slot >> 1, h = slot & 1;
  int t  = frag >> 1, p = frag & 1;        // t: (k1,k2,k3) tap; p: K-pair
  int lh = lane >> 4, m = lane & 15;       // m = Cout row
  // 16-bit A 16x32 layout (contiguous K to match B): K = lh*16 + 2v + h
  int k  = lh * 16 + v * 2 + h;
  int tau = p * 2 + (k >> 4);              // k4 tap: 0,1 (p=0) / 2,3 (p=1; 3=pad)
  int ci  = k & 15;
  float val = 0.f;
  if (tau < 3 && m < coutR && ci < cinR) {
    int k1 = t / 9, k2 = (t / 3) % 3, k3 = t % 3;
    val = w[((((m * cinR + ci) * 3 + k1) * 3 + k2) * 3 + k3) * 3 + tau];
  }
  wp[idx] = f32_to_bf16(val);
}

// ---------------------------------------------------------------------------
// Pack input x (f32, [B,1,32,32,32,32]) into channel-last padded bf16
// activations [bd1][d2][d3][d4][16]. transposed=1 reads x with the
// (d1,d2)<->(d3,d4) permutation for the symmetric branch.
// ---------------------------------------------------------------------------
__global__ void nc_pack_x(const float* __restrict__ x,
                          unsigned short* __restrict__ dst, int transposed) {
  long idx = (long)blockIdx.x * blockDim.x + threadIdx.x;
  if (idx >= NPOS) return;
  int d4 = idx & 31; long r = idx >> 5;
  int d3 = r & 31; r >>= 5;
  int d2 = r & 31; r >>= 5;
  int bd1 = (int)r, b = bd1 >> 5, d1 = bd1 & 31;
  long src = transposed
      ? (((((long)b * 32 + d3) * 32 + d4) * 32 + d1) * 32 + d2)
      : idx;
  uint4 lo = {0u, 0u, 0u, 0u}, hi = {0u, 0u, 0u, 0u};
  lo.x = (unsigned)f32_to_bf16(x[src]);     // ci=0 real, ci=1..15 zero-pad
  uint4* o = (uint4*)&dst[idx * CPAD];
  o[0] = lo; o[1] = hi;
}

// ---------------------------------------------------------------------------
// One 4D conv layer as implicit-im2col WMMA GEMM with register N-blocking
// and LDS d3-blocking.
// Block = 64 threads (2 waves); block covers (bd1, d2, d3b..d3b+1).
// Wave w handles d3 = d3b + w and computes the FULL d4 row (two 16-wide N
// tiles, two f32 accumulators) so every A fragment load feeds 2 WMMAs.
// LDS patch: 36 rows (3*3 taps x 4 s3-halo) x 34 d4-halo x 16 ch bf16 = 38 KB.
// finalMode: 0 = store bf16 activations (ReLU); 1 = branch-1 f32 '=' store;
//            2 = branch-2 f32 '+=' store with output transpose.
// ---------------------------------------------------------------------------
__global__ __launch_bounds__(64)
void nc_conv4d(const unsigned short* __restrict__ in,
               const unsigned short* __restrict__ wp,
               const float* __restrict__ bias,
               void* __restrict__ outv,
               int coutR, int finalMode) {
  const int d3b = (blockIdx.x & 15) << 1;   // first of 2 d3 positions
  const int d2  = blockIdx.x >> 4;
  const int bd1 = blockIdx.y;
  const int d1  = bd1 & 31;
  const int b32 = bd1 - d1;                 // b*32

  __shared__ __align__(32) uint4 patch4[PCHUNK];
  unsigned short* patch = (unsigned short*)patch4;

  const int tid = threadIdx.x;
  // Cooperative halo-patch fill: PCHUNK 16-byte chunks.
  // row = ((k1*3 + k2)*4 + j), j in 0..3 -> s3 = d3b + j - 1.
  for (int c = tid; c < PCHUNK; c += 64) {
    int row = c / 68, rem = c - row * 68;
    int d4p = rem >> 1, h8 = rem & 1;
    int j  = row & 3, rk = row >> 2;        // rk = k1*3 + k2
    int s1 = d1 + rk / 3 - 1;
    int s2 = d2 + rk % 3 - 1;
    int s3 = d3b + j - 1;
    int s4 = d4p - 1;
    uint4 val = {0u, 0u, 0u, 0u};
    if (((unsigned)s1 < 32u) && ((unsigned)s2 < 32u) &&
        ((unsigned)s3 < 32u) && ((unsigned)s4 < 32u)) {
      long off = (((((long)(b32 + s1) * 32 + s2) * 32 + s3) * 32 + s4)) * CPAD
               + (long)h8 * 8;
      val = *(const uint4*)&in[off];
    }
    patch4[c] = val;
  }
  __syncthreads();

  const int lane    = tid & 31;
  const int halfsel = lane >> 4;            // K half / Cout half selector
  const int n       = lane & 15;            // N column (d4 within tile)
  const int wid     = tid >> 5;             // wave -> which d3 of the pair

  const v16bf* wvp = (const v16bf*)wp;
  v8f acc0 = {};                            // d4 tile [0,16)
  v8f acc1 = {};                            // d4 tile [16,32)

#pragma unroll
  for (int t = 0; t < NTAP27; ++t) {
    const int dd1 = t / 9, dd2 = (t / 3) % 3, dd3 = t % 3;
    const int rowbase = ((dd1 * 3 + dd2) * 4 + (wid + dd3)) * 34 * CPAD;
#pragma unroll
    for (int p = 0; p < 2; ++p) {
      // p=0: K[0..15]->k4=0, K[16..31]->k4=1.  p=1: K[0..15]->k4=2, rest pad.
      const int off = p ? 2 : halfsel;
      v16bf af = wvp[(t * 2 + p) * 32 + lane];
      v16bf b0 = *(const v16bf*)&patch[rowbase + (n + off) * CPAD];
      v16bf b1 = *(const v16bf*)&patch[rowbase + (16 + n + off) * CPAD];
      acc0 = __builtin_amdgcn_wmma_f32_16x16x32_bf16(
          false, af, false, b0, (short)0, acc0, false, false);
      acc1 = __builtin_amdgcn_wmma_f32_16x16x32_bf16(
          false, af, false, b1, (short)0, acc1, false, false);
    }
  }

  const int d3 = d3b + wid;
  if (finalMode == 0) {
    // bf16 activation store: lane writes ci = 8*halfsel..+7 for 2 d4 columns.
    unsigned short* outb = (unsigned short*)outv;
    const long posbase = ((((long)bd1 * 32 + d2) * 32 + d3) * 32) * CPAD
                       + (long)halfsel * 8;
#pragma unroll
    for (int tile = 0; tile < 2; ++tile) {
      const v8f& a = tile ? acc1 : acc0;
      unsigned pk[4];
#pragma unroll
      for (int r = 0; r < 8; ++r) {
        int co = r + halfsel * 8;
        float v = a[r] + ((co < coutR) ? bias[co] : 0.f);
        v = v > 0.f ? v : 0.f;
        unsigned short bv = f32_to_bf16(v);
        if (r & 1) pk[r >> 1] |= ((unsigned)bv) << 16;
        else       pk[r >> 1]  = (unsigned)bv;
      }
      uint4 st; st.x = pk[0]; st.y = pk[1]; st.z = pk[2]; st.w = pk[3];
      *(uint4*)&outb[posbase + (long)(tile * 16 + n) * CPAD] = st;
    }
  } else {
    // Final layer: only cout=0 is real (lanes 0-15, accumulator row 0).
    if (halfsel == 0) {
      float v0 = acc0[0] + bias[0]; v0 = v0 > 0.f ? v0 : 0.f;
      float v1 = acc1[0] + bias[0]; v1 = v1 > 0.f ? v1 : 0.f;
      float* o = (float*)outv;
      if (finalMode == 1) {
        long pos = (((long)bd1 * 32 + d2) * 32 + d3) * 32;
        o[pos + n]      = v0;               // branch 1: overwrite (graph-safe)
        o[pos + n + 16] = v1;
      } else {
        int b = bd1 >> 5, d1l = bd1 & 31;   // branch 2: y += transpose(y_sym)
        long base = ((((long)b * 32 + d3) * 32) * 32 + d1l) * 32 + d2;
        o[base + (long)n * 1024]        += v0;   // d4' = n
        o[base + (long)(n + 16) * 1024] += v1;   // d4' = n+16
      }
    }
  }
}

// ---------------------------------------------------------------------------
// Workspace layout (requires ~269 MB):
//   [0, 128MB)      bufA  (bf16 activations, padded 16ch)
//   [128MB, 256MB)  bufB
//   [256MB, ...)    packed weights for the 3 layers (55296 B each)
// ---------------------------------------------------------------------------
extern "C" void kernel_launch(void* const* d_in, const int* in_sizes, int n_in,
                              void* d_out, int out_size, void* d_ws, size_t ws_size,
                              hipStream_t stream) {
  (void)in_sizes; (void)n_in; (void)out_size; (void)ws_size;
  const float* x  = (const float*)d_in[0];
  const float* w1 = (const float*)d_in[1];
  const float* b1 = (const float*)d_in[2];
  const float* w2 = (const float*)d_in[3];
  const float* b2 = (const float*)d_in[4];
  const float* w3 = (const float*)d_in[5];
  const float* b3 = (const float*)d_in[6];

  const size_t BUF = (size_t)NPOS * CPAD * sizeof(unsigned short); // 134217728
  unsigned short* bufA = (unsigned short*)d_ws;
  unsigned short* bufB = (unsigned short*)((char*)d_ws + BUF);
  unsigned short* wp1  = (unsigned short*)((char*)d_ws + 2 * BUF);
  unsigned short* wp2  = wp1 + FRAGS * FRAG_ELS;
  unsigned short* wp3  = wp2 + FRAGS * FRAG_ELS;

  // Weight pre-pack (tiny; re-done every call -> deterministic).
  dim3 wgrid((FRAGS * FRAG_ELS + 255) / 256);
  nc_pack_w<<<wgrid, 256, 0, stream>>>(w1, wp1, 10, 1);
  nc_pack_w<<<wgrid, 256, 0, stream>>>(w2, wp2, 10, 10);
  nc_pack_w<<<wgrid, 256, 0, stream>>>(w3, wp3, 1, 10);

  dim3 pgrid((unsigned)(NPOS / 256));
  dim3 cgrid(32 * 16, 128);   // (d2*16 + d3pair, bd1)

  for (int branch = 0; branch < 2; ++branch) {
    nc_pack_x<<<pgrid, 256, 0, stream>>>(x, bufA, branch);
    nc_conv4d<<<cgrid, 64, 0, stream>>>(bufA, wp1, b1, bufB, 10, 0);
    nc_conv4d<<<cgrid, 64, 0, stream>>>(bufB, wp2, b2, bufA, 10, 0);
    nc_conv4d<<<cgrid, 64, 0, stream>>>(bufA, wp3, b3, d_out, 1,
                                        branch == 0 ? 1 : 2);
  }
}
